// Weave_8160437862946
// MI455X (gfx1250) — compile-verified, matrix operand order checked by reading, not run on hardware
//
#include <hip/hip_runtime.h>
#include <hip/hip_bf16.h>

// ---------------------------------------------------------------------------
// Types for WMMA (CDNA5 gfx1250, wave32)
// ---------------------------------------------------------------------------
typedef __attribute__((ext_vector_type(16))) __bf16         v16bf;
typedef __attribute__((ext_vector_type(8)))  float          v8f;
typedef __attribute__((ext_vector_type(8)))  unsigned short ush8;
typedef __attribute__((ext_vector_type(16))) unsigned short ush16;

#define HD __device__ __forceinline__

HD unsigned short f2bf(float f) {
    unsigned int u = __builtin_bit_cast(unsigned int, f);
    u += 0x7FFFu + ((u >> 16) & 1u);         // round-to-nearest-even
    return (unsigned short)(u >> 16);
}

// Fragment load from an LDS tile stored row-major [rows][32] (bf16 bits).
// 16-bit A/B fragment layout (ISA 7.12.2): lane L -> row (L&15),
// K-base kb = (L&16)?8:0 ; element i -> K = kb + i + (i>=8 ? 8 : 0)
// => two contiguous 16-byte DS reads at k = kb and k = kb+16.
HD v16bf load_frag(const unsigned short* lds, int rowBase, int lane) {
    const int r  = rowBase + (lane & 15);
    const int kb = (lane & 16) ? 8 : 0;
    const ush8* p = (const ush8*)(lds + r * 32 + kb);
    ush8 lo = p[0];   // K = kb   .. kb+7
    ush8 hi = p[2];   // K = kb+16.. kb+23
    ush16 v = __builtin_shufflevector(lo, hi, 0,1,2,3,4,5,6,7,8,9,10,11,12,13,14,15);
    return __builtin_bit_cast(v16bf, v);
}

HD v8f wmma_bf16(v16bf a, v16bf b, v8f c) {
    return __builtin_amdgcn_wmma_f32_16x16x32_bf16(
        /*neg_a=*/false, a, /*neg_b=*/false, b,
        /*c_mod=*/(short)0, c, /*reuse_a=*/false, /*reuse_b=*/false);
}

HD void zero16(float* o) {
    #pragma unroll
    for (int i = 0; i < 16; i++) o[i] = 0.0f;
}
HD void f4copy(float* o, float4 a) { o[0]=a.x; o[1]=a.y; o[2]=a.z; o[3]=a.w; }

// ---------------------------------------------------------------------------
// A-matrix loaders: load 16 contiguous floats at (row, k).
// One row-guard per call, vectorized float4 (global_load_b128) loads.
// ---------------------------------------------------------------------------
struct LoadDirect {
    const float* A; int K; int M;
    HD void load16(int row, int k, float* o) const {
        if (row >= M) { zero16(o); return; }
        const float4* p = (const float4*)(A + (long)row * K + k);
        #pragma unroll
        for (int q = 0; q < 4; q++) f4copy(o + q * 4, p[q]);
    }
};
struct LoadConcat2 {                  // A = [A0 | A1], K0/K1 multiples of 16
    const float* A0; const float* A1; int K0; int K1; int M;
    HD void load16(int row, int k, float* o) const {
        if (row >= M) { zero16(o); return; }
        const float4* p = (k < K0)
            ? (const float4*)(A0 + (long)row * K0 + k)
            : (const float4*)(A1 + (long)row * K1 + (k - K0));
        #pragma unroll
        for (int q = 0; q < 4; q++) f4copy(o + q * 4, p[q]);
    }
};
struct LoadEdge {                     // A = [relu(L[s]+R[d]) | relu(R[s]+L[d]) | third]
    const float* left; const float* right; const float* third;
    const int* src; const int* dst; int M;
    HD void load16(int row, int k, float* o) const {
        if (row >= M) { zero16(o); return; }
        if (k < 512) {
            const int s = src[row], d = dst[row];
            const float4 *pa, *pb;
            if (k < 256) {
                pa = (const float4*)(left  + (long)s * 256 + k);
                pb = (const float4*)(right + (long)d * 256 + k);
            } else {
                pa = (const float4*)(right + (long)s * 256 + (k - 256));
                pb = (const float4*)(left  + (long)d * 256 + (k - 256));
            }
            #pragma unroll
            for (int q = 0; q < 4; q++) {
                float4 a = pa[q], b = pb[q];
                float v0 = a.x + b.x, v1 = a.y + b.y, v2 = a.z + b.z, v3 = a.w + b.w;
                o[q*4+0] = v0 > 0.f ? v0 : 0.f;
                o[q*4+1] = v1 > 0.f ? v1 : 0.f;
                o[q*4+2] = v2 > 0.f ? v2 : 0.f;
                o[q*4+3] = v3 > 0.f ? v3 : 0.f;
            }
        } else {
            const float4* p = (const float4*)(third + (long)row * 256 + (k - 512));
            #pragma unroll
            for (int q = 0; q < 4; q++) f4copy(o + q * 4, p[q]);
        }
    }
};

// ---------------------------------------------------------------------------
// Tiled WMMA GEMM: out(MxN) = act( A(MxK) @ Wt(NxK)^T + bias )
// Block 256 threads (8 waves). Tile 128(M) x 128(N), BK=32.
// Wave (wrow 0..3, wcol 0..1) owns a 32x64 patch = 8 wmma accumulators.
// ---------------------------------------------------------------------------
template <typename Loader>
__global__ __launch_bounds__(256)
void gemm_wmma(Loader ld, const unsigned short* __restrict__ Wt,
               const float* __restrict__ bias, float* __restrict__ out,
               int M, int N, int K, int act) {
    __shared__ unsigned short sA[128 * 32];   // 8 KB
    __shared__ unsigned short sB[128 * 32];   // 8 KB

    const int tid  = threadIdx.x;
    const int lane = tid & 31;
    const int wave = tid >> 5;
    const int wrow = wave >> 1, wcol = wave & 1;
    const int blockM = blockIdx.x * 128;
    const int blockN = blockIdx.y * 128;

    const v8f z = {0.f,0.f,0.f,0.f,0.f,0.f,0.f,0.f};
    v8f acc[8] = {z, z, z, z, z, z, z, z};

    union U8 { ush8 v; unsigned short s[8]; };

    // Per-thread staging coordinates (16 contiguous elements per thread)
    const int aRow = tid >> 1;             // 0..127
    const int aKK  = (tid & 1) * 16;       // 0 or 16

    for (int k0 = 0; k0 < K; k0 += 32) {
        // ---- stage A tile: 128x32, fp32 -> bf16 (vector loads, one guard) ----
        {
            float f[16];
            ld.load16(blockM + aRow, k0 + aKK, f);
            U8 v0, v1;
            #pragma unroll
            for (int i = 0; i < 8; i++) v0.s[i] = f2bf(f[i]);
            #pragma unroll
            for (int i = 0; i < 8; i++) v1.s[i] = f2bf(f[8 + i]);
            *(ush8*)(sA + aRow * 32 + aKK)     = v0.v;
            *(ush8*)(sA + aRow * 32 + aKK + 8) = v1.v;
        }
        // ---- stage B tile: 128x32 bf16 from pre-transposed weights ----
        {
            const long base = (long)(blockN + aRow) * K + k0 + aKK;
            ush8 w0 = *(const ush8*)(Wt + base);
            ush8 w1 = *(const ush8*)(Wt + base + 8);
            *(ush8*)(sB + aRow * 32 + aKK)     = w0;
            *(ush8*)(sB + aRow * 32 + aKK + 8) = w1;
        }
        __syncthreads();

        v16bf a0 = load_frag(sA, wrow * 32,      lane);
        v16bf a1 = load_frag(sA, wrow * 32 + 16, lane);
        v16bf b0 = load_frag(sB, wcol * 64,      lane);
        v16bf b1 = load_frag(sB, wcol * 64 + 16, lane);
        v16bf b2 = load_frag(sB, wcol * 64 + 32, lane);
        v16bf b3 = load_frag(sB, wcol * 64 + 48, lane);

        acc[0] = wmma_bf16(a0, b0, acc[0]);
        acc[1] = wmma_bf16(a0, b1, acc[1]);
        acc[2] = wmma_bf16(a0, b2, acc[2]);
        acc[3] = wmma_bf16(a0, b3, acc[3]);
        acc[4] = wmma_bf16(a1, b0, acc[4]);
        acc[5] = wmma_bf16(a1, b1, acc[5]);
        acc[6] = wmma_bf16(a1, b2, acc[6]);
        acc[7] = wmma_bf16(a1, b3, acc[7]);
        __syncthreads();
    }

    // ---- store: C/D layout = VGPR r -> row r (+8 for lanes 16..31), col = lane&15
    const int rbase = (lane & 16) ? 8 : 0;
    const int ncol  = lane & 15;
    #pragma unroll
    for (int i = 0; i < 2; i++) {
        #pragma unroll
        for (int j = 0; j < 4; j++) {
            const int t  = i * 4 + j;
            const int mt = blockM + wrow * 32 + i * 16 + rbase;
            const int nt = blockN + wcol * 64 + j * 16 + ncol;
            const float bv = bias[nt];
            #pragma unroll
            for (int r = 0; r < 8; r++) {
                const int m = mt + r;
                if (m < M) {
                    float v = acc[t][r] + bv;
                    if (act == 1)      v = v > 0.f ? v : 0.f;
                    else if (act == 2) v = tanhf(v);
                    out[(long)m * N + nt] = v;
                }
            }
        }
    }
}

// ---------------------------------------------------------------------------
// Weight convert + transpose: W(fin x fout, f32) -> Wt(fout x fin, bf16 bits)
// ---------------------------------------------------------------------------
__global__ void convw(const float* __restrict__ W, unsigned short* __restrict__ Wt,
                      int fin, int fout) {
    int idx = blockIdx.x * blockDim.x + threadIdx.x;
    if (idx >= fin * fout) return;
    int i = idx / fout, o = idx - i * fout;
    Wt[(long)o * fin + i] = f2bf(W[idx]);
}

// ---------------------------------------------------------------------------
// segment_sum: agg[dst[e]][c] += vals[e][c]  (H = 256, float4 per thread)
// dst is unsorted -> atomics are inherent here.
// ---------------------------------------------------------------------------
__global__ void segsum(const float4* __restrict__ vals, const int* __restrict__ dst,
                       float* __restrict__ agg, int E) {
    int idx = blockIdx.x * blockDim.x + threadIdx.x;   // over E*64 float4s
    if (idx >= E * 64) return;
    int e = idx >> 6, c4 = idx & 63;
    float4 v = vals[idx];
    float* a = agg + (long)dst[e] * 256 + c4 * 4;
    atomicAdd(a + 0, v.x);
    atomicAdd(a + 1, v.y);
    atomicAdd(a + 2, v.z);
    atomicAdd(a + 3, v.w);
}

// ---------------------------------------------------------------------------
// BatchNorm batch stats: one block per feature column (GF = 256)
// ---------------------------------------------------------------------------
__global__ __launch_bounds__(256)
void bn_stats(const float* __restrict__ h, float* __restrict__ mu,
              float* __restrict__ rsig, int V) {
    const int c = blockIdx.x;
    float s = 0.f, s2 = 0.f;
    for (int v = threadIdx.x; v < V; v += blockDim.x) {
        float x = h[(long)v * 256 + c];
        s += x; s2 += x * x;
    }
    __shared__ float rs[256], rq[256];
    rs[threadIdx.x] = s; rq[threadIdx.x] = s2;
    __syncthreads();
    for (int o = 128; o > 0; o >>= 1) {
        if ((int)threadIdx.x < o) { rs[threadIdx.x] += rs[threadIdx.x + o];
                                    rq[threadIdx.x] += rq[threadIdx.x + o]; }
        __syncthreads();
    }
    if (threadIdx.x == 0) {
        float m   = rs[0] / (float)V;
        float var = rq[0] / (float)V - m * m;
        mu[c]   = m;
        rsig[c] = rsqrtf(var + 1e-5f);
    }
}

// ---------------------------------------------------------------------------
// Fused BN-apply + Gaussian membership + per-graph segment reduction.
// graph_ids is SORTED, so each graph is a contiguous node range:
// one block per graph, binary-search bounds (uniform/SALU), thread c owns
// feature column c and keeps the 11 bin accumulators in registers.
// No atomics, no init memset needed (every (g, c, j) written exactly once).
// ---------------------------------------------------------------------------
__constant__ float c_means[11]  = {-1.645f,-1.080f,-0.739f,-0.468f,-0.228f,0.0f,
                                    0.228f, 0.468f, 0.739f, 1.080f, 1.645f};
__constant__ float c_invstd[11] = {3.5335689f,5.8823529f,7.4626866f,8.4745763f,
                                   8.7719298f,8.7719298f,8.7719298f,8.4745763f,
                                   7.4626866f,5.8823529f,3.5335689f};
__constant__ float c_coef[11]   = {1.4096926f,2.3467203f,2.9771795f,3.3808663f,
                                   3.4994932f,3.4994932f,3.4994932f,3.3808663f,
                                   2.9771795f,2.3467203f,1.4096926f};

__global__ __launch_bounds__(256)
void graph_hist(const float* __restrict__ h, const int* __restrict__ gid,
                const float* __restrict__ mu, const float* __restrict__ rsig,
                const float* __restrict__ gamma, const float* __restrict__ beta,
                float* __restrict__ gfeats, int V) {
    const int g = blockIdx.x;
    // lower_bound(g) and lower_bound(g+1) over sorted gid[0..V)
    int a = 0, b = V;
    while (a < b) { int m = (a + b) >> 1; if (gid[m] < g)     a = m + 1; else b = m; }
    const int lo = a;
    b = V;
    while (a < b) { int m = (a + b) >> 1; if (gid[m] < g + 1) a = m + 1; else b = m; }
    const int hi = a;

    const int c = threadIdx.x;                 // feature column 0..255
    const float gm = gamma[c], bt = beta[c], mC = mu[c], rC = rsig[c];

    float accv[11];
    #pragma unroll
    for (int j = 0; j < 11; j++) accv[j] = 0.f;

    for (int v = lo; v < hi; v++) {
        float x = gm * (h[(long)v * 256 + c] - mC) * rC + bt;
        float p[11], sum = 0.f;
        #pragma unroll
        for (int j = 0; j < 11; j++) {
            float zz = (x - c_means[j]) * c_invstd[j];
            float pj = __expf(-0.5f * zz * zz) * c_coef[j];
            p[j] = pj; sum += pj;
        }
        float inv = 1.0f / sum;
        #pragma unroll
        for (int j = 0; j < 11; j++) accv[j] += p[j] * inv;
    }

    float* gp = gfeats + (long)g * 2816 + c * 11;
    #pragma unroll
    for (int j = 0; j < 11; j++) gp[j] = accv[j];
}

// ---------------------------------------------------------------------------
// Host side
// ---------------------------------------------------------------------------
static inline void* wsCarve(void* ws, size_t& off, size_t bytes) {
    void* p = (char*)ws + off;
    off += (bytes + 255) & ~(size_t)255;
    return p;
}

extern "C" void kernel_launch(void* const* d_in, const int* in_sizes, int n_in,
                              void* d_out, int out_size, void* d_ws, size_t ws_size,
                              hipStream_t stream) {
    (void)in_sizes; (void)n_in; (void)out_size; (void)ws_size;
    const int V = 50000, E = 200000, Gn = 2000, H = 256;

    const float* node_feats = (const float*)d_in[0];
    const float* edge_feats = (const float*)d_in[1];
    const int*   src        = (const int*)d_in[2];
    const int*   dst        = (const int*)d_in[3];
    const int*   graph_ids  = (const int*)d_in[4];

    const float* w_n2n0  = (const float*)d_in[5];  const float* b_n2n0  = (const float*)d_in[6];
    const float* w_e2n0  = (const float*)d_in[7];  const float* b_e2n0  = (const float*)d_in[8];
    const float* w_updn0 = (const float*)d_in[9];  const float* b_updn0 = (const float*)d_in[10];
    const float* w_ln2e0 = (const float*)d_in[11]; const float* b_ln2e0 = (const float*)d_in[12];
    const float* w_rn2e0 = (const float*)d_in[13]; const float* b_rn2e0 = (const float*)d_in[14];
    const float* w_e2e0  = (const float*)d_in[15]; const float* b_e2e0  = (const float*)d_in[16];
    const float* w_upde0 = (const float*)d_in[17]; const float* b_upde0 = (const float*)d_in[18];
    const float* w_n2n1  = (const float*)d_in[19]; const float* b_n2n1  = (const float*)d_in[20];
    const float* w_e2n1  = (const float*)d_in[21]; const float* b_e2n1  = (const float*)d_in[22];
    const float* w_updn1 = (const float*)d_in[23]; const float* b_updn1 = (const float*)d_in[24];
    const float* w_n2g   = (const float*)d_in[25]; const float* b_n2g   = (const float*)d_in[26];
    const float* bn_g    = (const float*)d_in[27]; const float* bn_b    = (const float*)d_in[28];
    const float* w_out   = (const float*)d_in[29]; const float* b_out   = (const float*)d_in[30];

    // ---- workspace carve ----
    size_t off = 0;
    unsigned short* t_n2n0  = (unsigned short*)wsCarve(d_ws, off, (size_t)256*128*2);
    unsigned short* t_e2n0  = (unsigned short*)wsCarve(d_ws, off, (size_t)256*64*2);
    unsigned short* t_updn0 = (unsigned short*)wsCarve(d_ws, off, (size_t)256*512*2);
    unsigned short* t_ln2e0 = (unsigned short*)wsCarve(d_ws, off, (size_t)256*128*2);
    unsigned short* t_rn2e0 = (unsigned short*)wsCarve(d_ws, off, (size_t)256*128*2);
    unsigned short* t_e2e0  = (unsigned short*)wsCarve(d_ws, off, (size_t)256*64*2);
    unsigned short* t_upde0 = (unsigned short*)wsCarve(d_ws, off, (size_t)256*768*2);
    unsigned short* t_n2n1  = (unsigned short*)wsCarve(d_ws, off, (size_t)256*256*2);
    unsigned short* t_e2n1  = (unsigned short*)wsCarve(d_ws, off, (size_t)256*256*2);
    unsigned short* t_updn1 = (unsigned short*)wsCarve(d_ws, off, (size_t)256*512*2);
    unsigned short* t_n2g   = (unsigned short*)wsCarve(d_ws, off, (size_t)256*256*2);
    unsigned short* t_out   = (unsigned short*)wsCarve(d_ws, off, (size_t)256*2816*2);

    const size_t VHb = (size_t)V * H * sizeof(float);
    const size_t EHb = (size_t)E * H * sizeof(float);
    float* P0     = (float*)wsCarve(d_ws, off, VHb);   // nn_h -> left -> nn_h2 -> h
    float* P1     = (float*)wsCarve(d_ws, off, VHb);   // agg  -> right -> agg2
    float* P2     = (float*)wsCarve(d_ws, off, VHb);   // n1   -> n2
    float* bufE0  = (float*)wsCarve(d_ws, off, EHb);   // e2n0 -> e2e0(third) -> e2n1
    float* bufE1  = (float*)wsCarve(d_ws, off, EHb);   // e1
    float* gfeats = (float*)wsCarve(d_ws, off, (size_t)Gn * 2816 * sizeof(float));
    float* mu     = (float*)wsCarve(d_ws, off, 256 * sizeof(float));
    float* rsig   = (float*)wsCarve(d_ws, off, 256 * sizeof(float));

    // ---- convert + transpose all weights to bf16 [N][K] ----
    struct { const float* w; unsigned short* t; int fin; } cw[12] = {
        {w_n2n0,t_n2n0,128},{w_e2n0,t_e2n0,64},{w_updn0,t_updn0,512},
        {w_ln2e0,t_ln2e0,128},{w_rn2e0,t_rn2e0,128},{w_e2e0,t_e2e0,64},
        {w_upde0,t_upde0,768},{w_n2n1,t_n2n1,256},{w_e2n1,t_e2n1,256},
        {w_updn1,t_updn1,512},{w_n2g,t_n2g,256},{w_out,t_out,2816}};
    for (int i = 0; i < 12; i++) {
        int tot = cw[i].fin * 256;
        convw<<<(tot + 255) / 256, 256, 0, stream>>>(cw[i].w, cw[i].t, cw[i].fin, 256);
    }

    const dim3 gridV((V + 127) / 128, H / 128);
    const dim3 gridE((E + 127) / 128, H / 128);
    const dim3 gridG((Gn + 127) / 128, H / 128);

    // ---- Weave layer 0 ----
    gemm_wmma<<<gridV, 256, 0, stream>>>(LoadDirect{node_feats, 128, V},
        t_n2n0, b_n2n0, P0, V, H, 128, 1);                               // nn_h
    gemm_wmma<<<gridE, 256, 0, stream>>>(LoadDirect{edge_feats, 64, E},
        t_e2n0, b_e2n0, bufE0, E, H, 64, 1);                             // e2n
    hipMemsetAsync(P1, 0, VHb, stream);
    segsum<<<(E * 64) / 256, 256, 0, stream>>>((const float4*)bufE0, dst, P1, E);
    gemm_wmma<<<gridV, 256, 0, stream>>>(LoadConcat2{P0, P1, 256, 256, V},
        t_updn0, b_updn0, P2, V, H, 512, 1);                             // n1
    gemm_wmma<<<gridV, 256, 0, stream>>>(LoadDirect{node_feats, 128, V},
        t_ln2e0, b_ln2e0, P0, V, H, 128, 0);                             // left
    gemm_wmma<<<gridV, 256, 0, stream>>>(LoadDirect{node_feats, 128, V},
        t_rn2e0, b_rn2e0, P1, V, H, 128, 0);                             // right
    gemm_wmma<<<gridE, 256, 0, stream>>>(LoadDirect{edge_feats, 64, E},
        t_e2e0, b_e2e0, bufE0, E, H, 64, 1);                             // third
    gemm_wmma<<<gridE, 256, 0, stream>>>(LoadEdge{P0, P1, bufE0, src, dst, E},
        t_upde0, b_upde0, bufE1, E, H, 768, 1);                          // e1

    // ---- Weave layer 1 (node only) ----
    gemm_wmma<<<gridV, 256, 0, stream>>>(LoadDirect{P2, 256, V},
        t_n2n1, b_n2n1, P0, V, H, 256, 1);                               // nn_h2
    gemm_wmma<<<gridE, 256, 0, stream>>>(LoadDirect{bufE1, 256, E},
        t_e2n1, b_e2n1, bufE0, E, H, 256, 1);                            // e2n (layer1)
    hipMemsetAsync(P1, 0, VHb, stream);
    segsum<<<(E * 64) / 256, 256, 0, stream>>>((const float4*)bufE0, dst, P1, E);
    gemm_wmma<<<gridV, 256, 0, stream>>>(LoadConcat2{P0, P1, 256, 256, V},
        t_updn1, b_updn1, P2, V, H, 512, 1);                             // n2

    // ---- node_to_graph: tanh(n2 @ w_n2g + b), then BN stats ----
    gemm_wmma<<<gridV, 256, 0, stream>>>(LoadDirect{P2, 256, V},
        t_n2g, b_n2g, P0, V, H, 256, 2);                                 // h
    bn_stats<<<256, 256, 0, stream>>>(P0, mu, rsig, V);

    // ---- fused BN-apply + gaussian membership + per-graph reduction ----
    graph_hist<<<Gn, 256, 0, stream>>>(P0, graph_ids, mu, rsig,
                                       bn_g, bn_b, gfeats, V);

    // ---- final: tanh(g_feats @ w_out + b_out) -> d_out (2000 x 256) ----
    gemm_wmma<<<gridG, 256, 0, stream>>>(LoadDirect{gfeats, 2816, Gn},
        t_out, b_out, (float*)d_out, Gn, H, 2816, 2);
}
